// cov_64939905516357
// MI455X (gfx1250) — compile-verified, matrix operand order checked by reading, not run on hardware
//
#include <hip/hip_runtime.h>

typedef __attribute__((ext_vector_type(16))) _Float16 v16h;
typedef __attribute__((ext_vector_type(8)))  _Float16 v8h;
typedef __attribute__((ext_vector_type(4)))  _Float16 v4h;
typedef __attribute__((ext_vector_type(8)))  float    v8f;
typedef __attribute__((ext_vector_type(4)))  int      v4i;

#define C_DIM 640
#define HW    441
#define NCLS  10
#define SHOTS 5
#define KTOT  (SHOTS*HW)   // 2205
#define KPAD  2208         // 69 * 32
#define DSTRIDE 512        // padded column stride of d-partial buffer

// CDNA5 async global->LDS copy support (probe via __has_builtin; fall back to
// a manual VGPR round-trip copy if the toolchain lacks the builtin).
#if defined(__AMDGCN__) && __has_builtin(__builtin_amdgcn_global_load_async_to_lds_b128)
#define HAVE_ASYNC_LDS 1
#else
#define HAVE_ASYNC_LDS 0
#endif

#if HAVE_ASYNC_LDS
// Param types per hipcc diagnostic: pointee is int4 in address_space(1) for the
// global side (printed as "int __device__ *"), AS(3) for the LDS side.
__device__ __forceinline__ void async_copy_b128(const void* g, void* l) {
    __builtin_amdgcn_global_load_async_to_lds_b128(
        (__attribute__((address_space(1))) v4i*)(v4i*)g,
        (__attribute__((address_space(3))) v4i*)(v4i*)l,
        /*offset=*/0, /*cpol=*/0);
}
#endif

__device__ __forceinline__ void wait_async0() {
#if defined(__AMDGCN__)
#if __has_builtin(__builtin_amdgcn_s_wait_asynccnt)
    __builtin_amdgcn_s_wait_asynccnt(0);
#else
    asm volatile("s_wait_asynccnt 0x0" ::: "memory");
#endif
#endif
}

// Build a 16x32-f16 WMMA operand fragment from LDS: two 16B runs of 8 halves.
// K(j,lane) = (j<8?0:16) + 8*(lane>>4) + (j%8)  -> runs at +0 and +16 halves.
__device__ __forceinline__ v16h ldfrag(const _Float16* p) {
    v8h lo = *(const v8h*)p;
    v8h hi = *(const v8h*)(p + 16);
    return __builtin_shufflevector(lo, hi, 0,1,2,3,4,5,6,7,8,9,10,11,12,13,14,15);
}

// ---------------------------------------------------------------------------
// Kernel A: cov[n] = ys ys^T / (S*hw - 1), output f16.
// grid (25, 10) : 5x5 tiles of 128x128 per class. block 256 = 8 waves.
// wave w: rows (w>>2)*64, cols (w&3)*32 -> 4x2 register tile, 8 WMMA / k-step.
// ---------------------------------------------------------------------------
__global__ __launch_bounds__(256) void cov_kernel(const float* __restrict__ y,
                                                  _Float16* __restrict__ covh)
{
    __shared__ __attribute__((aligned(16))) _Float16 Ash[128*40]; // stride 40 halves = 80B (16B mult)
    __shared__ __attribute__((aligned(16))) _Float16 Bsh[128*40];
    const int t    = threadIdx.x;
    const int lane = t & 31;
    const int w    = t >> 5;
    const int cls  = blockIdx.y;
    const int tm5  = blockIdx.x / 5, tn5 = blockIdx.x % 5;
    const int mbase = (w >> 2) * 64;
    const int nbase = (w & 3) * 32;
    const int l16 = lane & 15, lh = lane >> 4;

    v8f acc[4][2] = {};
    const int ybase = cls * SHOTS * C_DIM * HW;

    for (int k0 = 0; k0 < KPAD; k0 += 32) {
        // stage 128x32 f32->f16 chunks of ys rows for A and B tiles
        #pragma unroll 4
        for (int it = 0; it < 16; ++it) {
            int q   = t + it * 256;          // 0..4095
            int row = q >> 5, kk = q & 31;
            int i   = k0 + kk;
            float va = 0.f, vb = 0.f;
            if (i < KTOT) {
                int s = i / HW, h = i - s * HW;
                int off = ybase + s * C_DIM * HW + h;
                va = y[off + (tm5 * 128 + row) * HW];
                vb = y[off + (tn5 * 128 + row) * HW];
            }
            Ash[row * 40 + kk] = (_Float16)va;
            Bsh[row * 40 + kk] = (_Float16)vb;
        }
        __syncthreads();

        v16h a[4], b[2];
        #pragma unroll
        for (int mt = 0; mt < 4; ++mt)
            a[mt] = ldfrag(Ash + (mbase + mt * 16 + l16) * 40 + 8 * lh);
        #pragma unroll
        for (int nt = 0; nt < 2; ++nt)
            b[nt] = ldfrag(Bsh + (nbase + nt * 16 + l16) * 40 + 8 * lh);

        #pragma unroll
        for (int mt = 0; mt < 4; ++mt)
            #pragma unroll
            for (int nt = 0; nt < 2; ++nt)
                acc[mt][nt] = __builtin_amdgcn_wmma_f32_16x16x32_f16(
                    false, a[mt], false, b[nt], (short)0, acc[mt][nt], false, false);
        __syncthreads();
    }

    const float scale = 1.f / (float)(KTOT - 1);
    #pragma unroll
    for (int mt = 0; mt < 4; ++mt)
        #pragma unroll
        for (int nt = 0; nt < 2; ++nt)
            #pragma unroll
            for (int r = 0; r < 8; ++r) {
                int M = tm5 * 128 + mbase + mt * 16 + r + 8 * lh;  // C/D: M = r + 8*(lane>>4)
                int N = tn5 * 128 + nbase + nt * 16 + l16;         //       N = lane%16
                covh[(cls * C_DIM + M) * C_DIM + N] = (_Float16)(acc[mt][nt][r] * scale);
            }
}

// ---------------------------------------------------------------------------
// Kernel B: d[b,n,h] = x_col^T Cov_n x_col, fused over all 10 classes.
// grid (4, 256): 128 padded columns (h) per WG, one query b. block 256 = 8 waves.
// wave w: column group ng = w&3 (32 cols, 2 n-tiles), c-half cg = w>>2 (320 rows).
// X block stays resident in LDS (f16, transposed [col][c]); cov row-blocks are
// copied global->LDS with CDNA5 async-to-LDS b128 (ASYNCcnt), no VGPR traffic.
// 2x2 register tile -> 4 WMMA per 8 ds_load_b128.
// ---------------------------------------------------------------------------
__global__ __launch_bounds__(256) void quad_kernel(const float* __restrict__ x,
                                                   const _Float16* __restrict__ covh,
                                                   float* __restrict__ dbuf)
{
    __shared__ __attribute__((aligned(16))) _Float16 xsh[128*648]; // [col][c], stride 648
    __shared__ __attribute__((aligned(16))) _Float16 csh[64*648];  // 64 cov rows
    const int t    = threadIdx.x;
    const int lane = t & 31;
    const int w    = t >> 5;
    const int b    = blockIdx.y;
    const int h0   = blockIdx.x * 128;
    const int l16  = lane & 15, lh = lane >> 4;
    const int ng   = w & 3;    // 32-column group
    const int cg   = w >> 2;   // 320-row half of C

    // stage x[b, :, h0:h0+128] -> f16 transposed; zero-pad h >= 441
    for (int it = 0; it < 320; ++it) {
        int q = t + it * 256;            // 0..81919
        int c = q >> 7, j = q & 127;
        int h = h0 + j;
        float v = (h < HW) ? x[(b * C_DIM + c) * HW + h] : 0.f;
        xsh[j * 648 + c] = (_Float16)v;
    }
    __syncthreads();

    for (int cls = 0; cls < NCLS; ++cls) {
        float dacc[2] = {0.f, 0.f};
        for (int ctl = 0; ctl < 10; ++ctl) {
            // stage 64 cov rows: rr -> global row (rr>>5)*320 + ctl*32 + (rr&31)
#if HAVE_ASYNC_LDS
            #pragma unroll 4
            for (int it = 0; it < 20; ++it) {
                int u   = t + it * 256;       // 0..5119 (b128 units; 80 per row)
                int rr  = u / 80, c16 = u % 80;
                int g   = ((rr >> 5) * 320) + ctl * 32 + (rr & 31);
                async_copy_b128(covh + (cls * C_DIM + g) * C_DIM + c16 * 8,
                                csh + rr * 648 + c16 * 8);
            }
            wait_async0();
#else
            #pragma unroll 4
            for (int it = 0; it < 40; ++it) {
                int u  = t + it * 256;        // 0..10239 (v4h units)
                int rr = u / 160, c4 = (u % 160) * 4;
                int g  = ((rr >> 5) * 320) + ctl * 32 + (rr & 31);
                *(v4h*)(csh + rr * 648 + c4) =
                    *(const v4h*)(covh + (cls * C_DIM + g) * C_DIM + c4);
            }
#endif
            __syncthreads();

            v8f T[2][2] = {};
            for (int e0 = 0; e0 < C_DIM; e0 += 32) {
                v16h a[2], bf[2];
                #pragma unroll
                for (int mt = 0; mt < 2; ++mt)
                    a[mt] = ldfrag(csh + (cg * 32 + mt * 16 + l16) * 648 + e0 + 8 * lh);
                #pragma unroll
                for (int nt = 0; nt < 2; ++nt)
                    bf[nt] = ldfrag(xsh + (ng * 32 + nt * 16 + l16) * 648 + e0 + 8 * lh);
                #pragma unroll
                for (int mt = 0; mt < 2; ++mt)
                    #pragma unroll
                    for (int nt = 0; nt < 2; ++nt)
                        T[mt][nt] = __builtin_amdgcn_wmma_f32_16x16x32_f16(
                            false, a[mt], false, bf[nt], (short)0, T[mt][nt], false, false);
            }

            // d += x[c,col] * (Cov x)[c,col] for this 32-row slab
            #pragma unroll
            for (int nt = 0; nt < 2; ++nt) {
                int col = ng * 32 + nt * 16 + l16;
                #pragma unroll
                for (int mt = 0; mt < 2; ++mt) {
                    int cb = cg * 320 + ctl * 32 + mt * 16 + 8 * lh;  // rows r+8*lh of T
                    v8h xv = *(const v8h*)(xsh + col * 648 + cb);
                    float s = 0.f;
                    #pragma unroll
                    for (int r = 0; r < 8; ++r) s += T[mt][nt][r] * (float)xv[r];
                    dacc[nt] += s;
                }
            }
            __syncthreads();
        }
        // combine the two M-half lanes (lane l and l+16 hold the same column)
        #pragma unroll
        for (int nt = 0; nt < 2; ++nt) {
            float tot = dacc[nt] + __shfl_xor(dacc[nt], 16, 32);
            if (lane < 16) {
                int col = h0 + ng * 32 + nt * 16 + l16;   // 0..511 (padded)
                dbuf[((b * NCLS + cls) * 2 + cg) * DSTRIDE + col] = tot;
            }
        }
    }
}

// ---------------------------------------------------------------------------
// Kernel C: out[b,n] = b441 + sum_h leaky(d0+d1, 0.2) * w441[h].  320 blocks,
// one wave per (b,n) output; deterministic wave reduction.
// ---------------------------------------------------------------------------
__global__ __launch_bounds__(256) void out_kernel(const float* __restrict__ dbuf,
                                                  const float* __restrict__ w441,
                                                  const float* __restrict__ b441,
                                                  float* __restrict__ out)
{
    const int t = threadIdx.x, lane = t & 31, w = t >> 5;
    const int g = blockIdx.x * 8 + w;          // 0..2559
    const float* d0 = dbuf + (size_t)g * 2 * DSTRIDE;
    const float* d1 = d0 + DSTRIDE;
    float acc = 0.f;
    for (int h = lane; h < HW; h += 32) {
        float v = d0[h] + d1[h];
        float a = v > 0.f ? v : 0.2f * v;
        acc += a * w441[h];
    }
    acc += __shfl_xor(acc, 16, 32);
    acc += __shfl_xor(acc, 8, 32);
    acc += __shfl_xor(acc, 4, 32);
    acc += __shfl_xor(acc, 2, 32);
    acc += __shfl_xor(acc, 1, 32);
    if (lane == 0) out[g] = acc + b441[0];
}

extern "C" void kernel_launch(void* const* d_in, const int* in_sizes, int n_in,
                              void* d_out, int out_size, void* d_ws, size_t ws_size,
                              hipStream_t stream) {
    (void)in_sizes; (void)n_in; (void)out_size; (void)ws_size;
    const float* x    = (const float*)d_in[0];   // [256,640,441]
    const float* y    = (const float*)d_in[1];   // [10,5,640,441]
    const float* w441 = (const float*)d_in[2];   // [441]
    const float* b441 = (const float*)d_in[3];   // [1]
    float* out = (float*)d_out;                  // [256,10] f32

    _Float16* covh = (_Float16*)d_ws;                                  // 8,192,000 B
    float*    dbuf = (float*)((char*)d_ws + (size_t)NCLS*C_DIM*C_DIM*sizeof(_Float16));
    // dbuf: [256][10][2][512] f32 = 10,485,760 B  (total ws use ~18.7 MB)

    cov_kernel <<<dim3(25, NCLS), 256, 0, stream>>>(y, covh);
    quad_kernel<<<dim3(4, 256),   256, 0, stream>>>(x, covh, dbuf);
    out_kernel <<<320,            256, 0, stream>>>(dbuf, w441, b441, out);
}
